// Encoder_21328807592379
// MI455X (gfx1250) — matrix operand order repaired and checked by reading, not compiled, not run on hardware
//
#include <hip/hip_runtime.h>
#include <math.h>

typedef unsigned short bf16_t;  // raw bf16 storage
typedef __attribute__((ext_vector_type(16))) __bf16 v16bf;
typedef __attribute__((ext_vector_type(8)))  float  v8f;

#define DEVI __device__ __forceinline__

DEVI float geluf(float x){ return 0.5f * x * (1.0f + erff(x * 0.70710678118654752f)); }
DEVI float sigf(float x){ return 1.0f / (1.0f + __expf(-x)); }
DEVI bf16_t f2bf(float f){
  unsigned u = __float_as_uint(f);
  unsigned r = (u + 0x7fffu + ((u >> 16) & 1u)) >> 16;  // round-to-nearest-even
  return (bf16_t)r;
}

// ---------------- elementwise utilities ----------------
__global__ void k_f2b(const float* __restrict__ X, bf16_t* __restrict__ Y, long long n){
  long long i = (long long)blockIdx.x * blockDim.x + threadIdx.x;
  if (i < n) Y[i] = f2bf(X[i]);
}
__global__ void k_add(const float* __restrict__ A, const float* __restrict__ B,
                      float* __restrict__ Y, long long n){
  long long i = (long long)blockIdx.x * blockDim.x + threadIdx.x;
  if (i < n) Y[i] = A[i] + B[i];
}
__global__ void k_copy(const float* __restrict__ A, float* __restrict__ Y, long long n){
  long long i = (long long)blockIdx.x * blockDim.x + threadIdx.x;
  if (i < n) Y[i] = A[i];
}

// ---------------- conv front-end ----------------
__global__ void k_conv1(const float* __restrict__ x, const float* __restrict__ w,
                        const float* __restrict__ b, float* __restrict__ y){
  long long idx = (long long)blockIdx.x * blockDim.x + threadIdx.x;
  if (idx >= (long long)8*72*4096) return;
  int l  = (int)(idx % 4096);
  int c  = (int)((idx / 4096) % 72);
  int bb = (int)(idx / (4096LL*72));
  int g = c >> 2;
  float v = b[c];
  #pragma unroll
  for (int k = 0; k < 3; ++k){
    int li = l + k - 1;
    if (li >= 0 && li < 4096) v += w[c*3 + k] * x[((size_t)bb*4096 + li)*18 + g];
  }
  y[idx] = v;
}
__global__ void k_c2pool(const float* __restrict__ in, const float* __restrict__ w,
                         const float* __restrict__ bi, const float* __restrict__ g,
                         const float* __restrict__ bt, const float* __restrict__ mn,
                         const float* __restrict__ vr, float* __restrict__ y){
  long long idx = (long long)blockIdx.x * blockDim.x + threadIdx.x;
  if (idx >= (long long)8*72*2048) return;
  int lp = (int)(idx % 2048);
  int c  = (int)((idx / 2048) % 72);
  int bb = (int)(idx / (2048LL*72));
  float inv = rsqrtf(vr[c] + 1e-5f);
  float best = -3.4e38f;
  for (int t = 0; t < 2; ++t){
    int l = 2*lp + t;
    const float* base = in + (size_t)bb*72*4096 + l;
    float v = bi[c];
    for (int j = 0; j < 72; ++j) v += w[c*72 + j] * base[(size_t)j*4096];
    v = (v - mn[c]) * inv * g[c] + bt[c];
    v = geluf(v);
    best = fmaxf(best, v);
  }
  y[idx] = best;
}
__global__ void k_c3(const float* __restrict__ in, const float* __restrict__ w,
                     const float* __restrict__ b, float* __restrict__ y){
  long long idx = (long long)blockIdx.x * blockDim.x + threadIdx.x;
  if (idx >= (long long)8*36*2048) return;
  int l  = (int)(idx % 2048);
  int c  = (int)((idx / 2048) % 36);
  int bb = (int)(idx / (2048LL*36));
  int g = c / 6;
  float v = b[c];
  #pragma unroll
  for (int j = 0; j < 12; ++j)
    v += w[c*12 + j] * in[((size_t)bb*72 + g*12 + j)*2048 + l];
  y[idx] = v;
}
__global__ void k_c4pool(const float* __restrict__ in, const float* __restrict__ w,
                         const float* __restrict__ bi, const float* __restrict__ g,
                         const float* __restrict__ bt, const float* __restrict__ mn,
                         const float* __restrict__ vr, float* __restrict__ y){
  long long idx = (long long)blockIdx.x * blockDim.x + threadIdx.x;
  if (idx >= (long long)8*36*1024) return;
  int lp = (int)(idx % 1024);
  int c  = (int)((idx / 1024) % 36);
  int bb = (int)(idx / (1024LL*36));
  float inv = rsqrtf(vr[c] + 1e-5f);
  float best = -3.4e38f;
  for (int t = 0; t < 2; ++t){
    int l = 2*lp + t;
    const float* base = in + (size_t)bb*36*2048 + l;
    float v = bi[c];
    #pragma unroll
    for (int j = 0; j < 36; ++j) v += w[c*36 + j] * base[(size_t)j*2048];
    v = (v - mn[c]) * inv * g[c] + bt[c];
    v = geluf(v);
    best = fmaxf(best, v);
  }
  y[idx] = best;
}
__global__ void k_c5t(const float* __restrict__ in, const float* __restrict__ w,
                      const float* __restrict__ b, float* __restrict__ y){
  long long idx = (long long)blockIdx.x * blockDim.x + threadIdx.x;
  if (idx >= (long long)8*1024*36) return;
  int c  = (int)(idx % 36);
  int l  = (int)((idx / 36) % 1024);
  int bb = (int)(idx / (36LL*1024));
  float v = b[c];
  for (int j = 0; j < 36; ++j){
    #pragma unroll
    for (int k = 0; k < 3; ++k){
      int li = l + k - 1;
      if (li >= 0 && li < 1024)
        v += w[(c*36 + j)*3 + k] * in[((size_t)bb*36 + j)*1024 + li];
    }
  }
  y[idx] = v;
}

// ---------------- generic fp32 linear ----------------
__global__ void k_linear(const float* __restrict__ X, const float* __restrict__ W,
                         const float* __restrict__ Bv, float* __restrict__ Y,
                         long long R, int K, int N, int act){
  long long idx = (long long)blockIdx.x * blockDim.x + threadIdx.x;
  long long tot = R * N;
  if (idx >= tot) return;
  int c = (int)(idx % N);
  long long r = idx / N;
  const float* xr = X + r * K;
  const float* wr = W + (size_t)c * K;
  float v = Bv ? Bv[c] : 0.0f;
  for (int k = 0; k < K; ++k) v += xr[k] * wr[k];
  if (act == 1) v = geluf(v);
  Y[idx] = v;
}

// ---------------- layernorm ----------------
__global__ void k_ln(const float* __restrict__ X, const float* __restrict__ g,
                     const float* __restrict__ b, float* __restrict__ Y,
                     long long rows, int D, float eps){
  long long r = (long long)blockIdx.x * blockDim.x + threadIdx.x;
  if (r >= rows) return;
  const float* x = X + r * D;
  float m = 0.f;
  for (int d = 0; d < D; ++d) m += x[d];
  m /= (float)D;
  float v = 0.f;
  for (int d = 0; d < D; ++d){ float t = x[d] - m; v += t*t; }
  v /= (float)D;
  float inv = rsqrtf(v + eps);
  float* y = Y + r * D;
  for (int d = 0; d < D; ++d) y[d] = g[d] * (x[d] - m) * inv + b[d];
}

// ---------------- pooling / upsample / pos-encoding ----------------
__global__ void k_avgpool(const float* __restrict__ X, float* __restrict__ Y,
                          int B, int L, int C, int k){
  int Lp = L / k;
  long long idx = (long long)blockIdx.x * blockDim.x + threadIdx.x;
  if (idx >= (long long)B*Lp*C) return;
  int c  = (int)(idx % C);
  int lp = (int)((idx / C) % Lp);
  int b  = (int)(idx / ((long long)C*Lp));
  float s = 0.f;
  for (int kk = 0; kk < k; ++kk) s += X[((size_t)b*L + lp*k + kk)*C + c];
  Y[idx] = s / (float)k;
}
__global__ void k_up2(const float* __restrict__ X, float* __restrict__ Y,
                      int B, int L, int C){
  int Lo = 2*L;
  long long idx = (long long)blockIdx.x * blockDim.x + threadIdx.x;
  if (idx >= (long long)B*Lo*C) return;
  int c = (int)(idx % C);
  int o = (int)((idx / C) % Lo);
  int b = (int)(idx / ((long long)C*Lo));
  float src = fminf(fmaxf((float)o*0.5f - 0.25f, 0.0f), (float)(L-1));
  int lo = (int)floorf(src);
  int hi = min(lo + 1, L - 1);
  float fr = src - (float)lo;
  Y[idx] = X[((size_t)b*L + lo)*C + c]*(1.0f - fr) + X[((size_t)b*L + hi)*C + c]*fr;
}
__global__ void k_posenc(float* __restrict__ X, int L, int D){
  long long idx = (long long)blockIdx.x * blockDim.x + threadIdx.x;
  if (idx >= (long long)8*L*D) return;
  int c = (int)(idx % D);
  int l = (int)((idx / D) % L);
  float i2 = (float)((c >> 1) << 1);
  float ang = (float)l * __expf(-(i2 / (float)D) * 9.2103403719761836f); // ln(10000)
  X[idx] += (c & 1) ? __cosf(ang) : __sinf(ang);
}

// ---------------- LSTM scan ----------------
__global__ void k_lstm(const float* __restrict__ pre, const float* __restrict__ whh,
                       const float* __restrict__ bhh, float* __restrict__ out,
                       int L, int rev){
  __shared__ float h[18], c[18], g[72];
  int b = blockIdx.x;
  int tid = threadIdx.x;
  if (tid < 18){ h[tid] = 0.f; c[tid] = 0.f; }
  __syncthreads();
  for (int step = 0; step < L; ++step){
    int t = rev ? (L - 1 - step) : step;
    float gv = pre[((size_t)b*L + t)*72 + tid] + bhh[tid];
    #pragma unroll
    for (int k = 0; k < 18; ++k) gv += whh[tid*18 + k] * h[k];
    g[tid] = gv;
    __syncthreads();
    if (tid < 18){
      float ii = sigf(g[tid]);
      float ff = sigf(g[18 + tid]);
      float gg = tanhf(g[36 + tid]);
      float oo = sigf(g[54 + tid]);
      float cn = ff * c[tid] + ii * gg;
      c[tid] = cn;
      float hn = oo * tanhf(cn);
      h[tid] = hn;
      out[((size_t)b*L + t)*36 + rev*18 + tid] = hn;
    }
    __syncthreads();
  }
}

// ---------------- online-softmax attention (small head dims, DSTA) ----------------
template<int HD>
__global__ void k_attn(const float* __restrict__ Q, const float* __restrict__ K,
                       const float* __restrict__ V, float* __restrict__ O,
                       int B, int Lq, int Lk, int nh, int D, float scale){
  long long idx = (long long)blockIdx.x * blockDim.x + threadIdx.x;
  long long tot = (long long)B * nh * Lq;
  if (idx >= tot) return;
  int i = (int)(idx % Lq);
  int hh = (int)((idx / Lq) % nh);
  int b = (int)(idx / ((long long)Lq * nh));
  const float* qr = Q + ((size_t)b*Lq + i)*D + hh*HD;
  float q[HD], acc[HD];
  #pragma unroll
  for (int d = 0; d < HD; ++d){ q[d] = qr[d] * scale; acc[d] = 0.f; }
  float m = -3.4e38f, s = 0.f;
  for (int j = 0; j < Lk; ++j){
    const float* kr = K + ((size_t)b*Lk + j)*D + hh*HD;
    float sc = 0.f;
    #pragma unroll
    for (int d = 0; d < HD; ++d) sc += q[d] * kr[d];
    float nm = fmaxf(m, sc);
    float corr = __expf(m - nm);
    float p = __expf(sc - nm);
    s = s * corr + p;
    const float* vr = V + ((size_t)b*Lk + j)*D + hh*HD;
    #pragma unroll
    for (int d = 0; d < HD; ++d) acc[d] = acc[d]*corr + p*vr[d];
    m = nm;
  }
  float inv = 1.0f / s;
  float* orow = O + ((size_t)b*Lq + i)*D + hh*HD;
  #pragma unroll
  for (int d = 0; d < HD; ++d) orow[d] = acc[d] * inv;
}

// ---------------- fragment loader: B-matrix (32xK16) from 16-bit row-major ----------------
// B[kk,n] layout: VGPR r holds kk = 2r (r<4) / 16+2(r-4) (r>=4) (+8 for lanes 16..31), n = lane&15
DEVI v16bf ldfragB(const bf16_t* base, int row, int stride, int kkbase, int hf){
  union { v16bf v; unsigned u[8]; } f;
  #pragma unroll
  for (int r = 0; r < 8; ++r)
    f.u[r] = *(const unsigned*)(base + (size_t)row*stride + kkbase + ((r < 4) ? (r<<1) : (16 + ((r-4)<<1))) + (hf<<3));
  return f.v;
}

// ---------------- WMMA flash attention: d=512, nh=8, hd=64, L=1024, B=8 ----------------
// wave = one 16-query block; block = 8 waves sharing (b,h) K/V tiles of 32 keys.
// K tile staged via async global->LDS; V staged transposed; P transposed via per-wave LDS.
__global__ __launch_bounds__(256) void k_fattn(const bf16_t* __restrict__ Qb,
                                               const bf16_t* __restrict__ Kb,
                                               const bf16_t* __restrict__ Vb,
                                               bf16_t* __restrict__ O, float scale){
  __shared__ bf16_t lds_k[32*64];       // [key][dim]
  __shared__ bf16_t lds_vt[64*32];      // [dim][key]
  __shared__ bf16_t lds_p[8][16*32];    // per-wave P tile [query][key]
  const int D = 512, L = 1024, HD = 64;
  int tid = threadIdx.x;
  int wave = tid >> 5, lane = tid & 31;
  int l15 = lane & 15, hf = lane >> 4;
  int blk = blockIdx.x;
  int qseg = blk & 7;
  int h    = (blk >> 3) & 7;
  int b    = blk >> 6;
  int qb   = qseg*8 + wave;            // query block 0..63
  // Q A-fragments for kk-chunks 0..31 and 32..63 (A layout: lane row = l15,
  // VGPR r -> K = 2r (r<4) / 16+2(r-4) (r>=4), +8 for lanes 16..31)
  union { v16bf v; unsigned u[8]; } fq[2];
  size_t baseQ = ((size_t)b*L + qb*16 + l15)*D + h*HD;
  #pragma unroll
  for (int c = 0; c < 2; ++c)
    #pragma unroll
    for (int r = 0; r < 8; ++r){
      int ka = ((r < 4) ? (r<<1) : (16 + ((r-4)<<1))) + (hf<<3);
      fq[c].u[r] = *(const unsigned*)(Qb + baseQ + c*32 + ka);
    }
  v8f oacc[4] = {};
  float rmax[8], rsum[8];
  #pragma unroll
  for (int r = 0; r < 8; ++r){ rmax[r] = -3.4e38f; rsum[r] = 0.f; }

  #pragma unroll 1
  for (int kb = 0; kb < 32; ++kb){
    __syncthreads();
    // ---- stage K tile (32 keys x 64 dims) via async load to LDS: 16B per lane ----
    {
      int key = tid >> 3, ch = tid & 7;
      const bf16_t* g = Kb + ((size_t)b*L + kb*32 + key)*D + h*HD + ch*8;
      unsigned loff = (unsigned)(uintptr_t)(lds_k + key*64 + ch*8);
      asm volatile("global_load_async_to_lds_b128 %0, %1, off"
                   :: "v"(loff), "v"(g) : "memory");
    }
    // ---- stage V tile transposed (dims x keys) ----
    #pragma unroll
    for (int i = 0; i < 8; ++i){
      int e = tid*8 + i;                 // 0..2047
      int key = e & 31, dim = e >> 5;
      lds_vt[dim*32 + key] = Vb[((size_t)b*L + kb*32 + key)*D + h*HD + dim];
    }
    asm volatile("s_wait_asynccnt 0" ::: "memory");
    __syncthreads();

    // ---- S = Q @ K^T : two 16x16 tiles (keys 0..15 / 16..31), kdim 64 = 2 wmma each ----
    v8f s0 = {}, s1 = {};
    s0 = __builtin_amdgcn_wmma_f32_16x16x32_bf16(false, fq[0].v, false,
           ldfragB(lds_k, l15,      64, 0,  hf), (short)0, s0, false, false);
    s0 = __builtin_amdgcn_wmma_f32_16x16x32_bf16(false, fq[1].v, false,
           ldfragB(lds_k, l15,      64, 32, hf), (short)0, s0, false, false);
    s1 = __builtin_amdgcn_wmma_f32_16x16x32_bf16(false, fq[0].v, false,
           ldfragB(lds_k, 16 + l15, 64, 0,  hf), (short)0, s1, false, false);
    s1 = __builtin_amdgcn_wmma_f32_16x16x32_bf16(false, fq[1].v, false,
           ldfragB(lds_k, 16 + l15, 64, 32, hf), (short)0, s1, false, false);

    // ---- online softmax per row (row = r + 8*hf lives across the 16-lane half) ----
    #pragma unroll
    for (int r = 0; r < 8; ++r){
      float v0 = s0[r]*scale, v1 = s1[r]*scale;
      float mx = fmaxf(v0, v1);
      #pragma unroll
      for (int off = 1; off < 16; off <<= 1) mx = fmaxf(mx, __shfl_xor(mx, off, 32));
      float nm = fmaxf(rmax[r], mx);
      float corr = __expf(rmax[r] - nm);
      rmax[r] = nm;
      float p0 = __expf(v0 - nm), p1 = __expf(v1 - nm);
      float ps = p0 + p1;
      #pragma unroll
      for (int off = 1; off < 16; off <<= 1) ps += __shfl_xor(ps, off, 32);
      rsum[r] = rsum[r]*corr + ps;
      #pragma unroll
      for (int t = 0; t < 4; ++t) oacc[t][r] *= corr;
      int row = r + 8*hf;
      lds_p[wave][row*32 + l15]      = f2bf(p0);
      lds_p[wave][row*32 + 16 + l15] = f2bf(p1);
    }
    asm volatile("s_wait_dscnt 0" ::: "memory");

    // ---- O += P @ V : P as A-fragment (16x32), V^T tiles as B ----
    union { v16bf v; unsigned u[8]; } fp;
    #pragma unroll
    for (int r = 0; r < 8; ++r){
      int ka = ((r < 4) ? (r<<1) : (16 + ((r-4)<<1))) + (hf<<3);
      fp.u[r] = *(const unsigned*)(&lds_p[wave][l15*32 + ka]);
    }
    #pragma unroll
    for (int t = 0; t < 4; ++t)
      oacc[t] = __builtin_amdgcn_wmma_f32_16x16x32_bf16(false, fp.v, false,
                  ldfragB(lds_vt, t*16 + l15, 32, 0, hf), (short)0, oacc[t], false, false);
  }

  // ---- epilogue: normalize, write bf16 ----
  #pragma unroll
  for (int t = 0; t < 4; ++t)
    #pragma unroll
    for (int r = 0; r < 8; ++r){
      int row = qb*16 + r + 8*hf;
      float val = oacc[t][r] / rsum[r];
      O[((size_t)b*L + row)*D + h*HD + t*16 + l15] = f2bf(val);
    }
}

// ---------------- WMMA bf16 GEMM: wave computes 16x64 strip (4 N-tiles, A reuse) ----------------
// Cf!=null -> fp32 out; else Ch bf16 out. act=1 -> exact GELU.
__global__ void k_gemm_bf16(const bf16_t* __restrict__ A, const bf16_t* __restrict__ W,
                            const float* __restrict__ bias, float* __restrict__ Cf,
                            bf16_t* __restrict__ Ch, int M, int N, int K, int act){
  int wave = (int)(((long long)blockIdx.x * blockDim.x + threadIdx.x) >> 5);
  int lane = threadIdx.x & 31;
  int tilesN = N >> 6;
  int tm = wave / tilesN;
  int tn = wave - tm * tilesN;
  if (tm >= (M >> 4)) return;
  int l15 = lane & 15;
  int hh  = lane >> 4;
  const bf16_t* Arow = A + (size_t)((tm << 4) + l15) * K;
  const bf16_t* W0   = W + (size_t)((tn << 6) + l15) * K;   // tile t row offset: t*16*K
  v8f acc[4] = {};
  union { v16bf v; unsigned u[8]; } fa, fb;
  for (int k0 = 0; k0 < K; k0 += 32){
    __builtin_prefetch(Arow + k0 + 128, 0, 1);
    __builtin_prefetch(W0   + k0 + 128, 0, 1);
    #pragma unroll
    for (int r = 0; r < 8; ++r){
      int ka = ((r < 4) ? (r << 1) : (16 + ((r - 4) << 1))) + (hh << 3);
      fa.u[r] = *(const unsigned int*)(Arow + k0 + ka);
    }
    #pragma unroll
    for (int t = 0; t < 4; ++t){
      #pragma unroll
      for (int r = 0; r < 8; ++r){
        int kb = ((r < 4) ? (r << 1) : (16 + ((r - 4) << 1))) + (hh << 3);
        fb.u[r] = *(const unsigned int*)(W0 + (size_t)t*16*K + k0 + kb);
      }
      acc[t] = __builtin_amdgcn_wmma_f32_16x16x32_bf16(false, fa.v, false, fb.v,
                                                       (short)0, acc[t], false, false);
    }
  }
  #pragma unroll
  for (int t = 0; t < 4; ++t){
    int ncol = (tn << 6) + (t << 4) + l15;
    float bv = bias ? bias[ncol] : 0.0f;
    #pragma unroll
    for (int r = 0; r < 8; ++r){
      float val = acc[t][r] + bv;
      if (act == 1) val = geluf(val);
      int mrow = (tm << 4) + r + (hh << 3);
      if (Cf) Cf[(size_t)mrow * N + ncol] = val;
      else    Ch[(size_t)mrow * N + ncol] = f2bf(val);
    }
  }
}

// ---------------- host side ----------------
struct MhaP { const float *iw, *ib, *ow, *ob; };
struct LayerP {
  const float *wq,*wqb,*wk,*wkb,*wv,*wvb,*wc,*wcb;
  const float *n1g,*n1b,*f1,*f1b,*f2,*f2bb,*n2g,*n2b;
};

static inline int nblk(long long n, int t){ return (int)((n + t - 1) / t); }

extern "C" void kernel_launch(void* const* d_in, const int* in_sizes, int n_in,
                              void* d_out, int out_size, void* d_ws, size_t ws_size,
                              hipStream_t stream){
  (void)in_sizes; (void)n_in; (void)out_size; (void)ws_size;
  int p = 0;
  auto F = [&](void)->const float* { return (const float*)d_in[p++]; };

  const float* x = F();
  const float *c1_w=F(), *c1_b=F(), *c2_w=F(), *c2_b=F();
  const float *bn1_g=F(), *bn1_b=F(), *bn1_m=F(), *bn1_v=F();
  const float *c3_w=F(), *c3_b=F(), *c4_w=F(), *c4_b=F();
  const float *bn2_g=F(), *bn2_b=F(), *bn2_m=F(), *bn2_v=F();
  const float *c5_w=F(), *c5_b=F();
  const float *wih_f=F(), *whh_f=F(), *bih_f=F(), *bhh_f=F();
  const float *wih_b=F(), *whh_b=F(), *bih_b=F(), *bhh_b=F();
  MhaP dl[6], dsm[6], dth[6], dc1[6], dc2[6];
  const float *dg[6], *db[6];
  for (int i = 0; i < 6; ++i){
    auto rd = [&](MhaP& m){ m.iw=F(); m.ib=F(); m.ow=F(); m.ob=F(); };
    rd(dl[i]); rd(dsm[i]); rd(dth[i]); rd(dc1[i]); rd(dc2[i]);
    dg[i]=F(); db[i]=F();
  }
  const float *proj_w=F(), *proj_b=F();
  const float *emb_w=F(),  *emb_b=F();
  LayerP lay[4];
  for (int i = 0; i < 4; ++i){
    lay[i].wq=F(); lay[i].wqb=F(); lay[i].wk=F(); lay[i].wkb=F();
    lay[i].wv=F(); lay[i].wvb=F(); lay[i].wc=F(); lay[i].wcb=F();
    lay[i].n1g=F(); lay[i].n1b=F();
    lay[i].f1=F(); lay[i].f1b=F(); lay[i].f2=F(); lay[i].f2bb=F();
    lay[i].n2g=F(); lay[i].n2b=F();
  }
  // src_mask: all ones -> no-op, ignored.

  char* wsP = (char*)d_ws;
  auto A32 = [&](long long e)->float*  { float*  r=(float*)wsP;  wsP += ((e*4 + 255)/256)*256; return r; };
  auto A16 = [&](long long e)->bf16_t* { bf16_t* r=(bf16_t*)wsP; wsP += ((e*2 + 255)/256)*256; return r; };

  const long long NSEQ = 8LL*1024*36;
  const long long NX   = 8LL*1024*512;
  float* bufA = A32(8LL*72*4096);
  float* bufB = A32(8LL*72*2048);
  float* bufC = A32(8LL*36*2048);
  float* bufD = A32(8LL*36*1024);
  float* hseq = A32(NSEQ);
  float* preF = A32(8LL*1024*72);
  float* preR = A32(8LL*1024*72);
  float* seq  = A32(NSEQ);
  float* bq=A32(NSEQ), *bk=A32(NSEQ), *bv=A32(NSEQ), *batt=A32(NSEQ);
  float* bLarge=A32(NSEQ), *bSmall=A32(NSEQ), *bThird=A32(NSEQ);
  float* bxp=A32(NSEQ), *bsu=A32(NSEQ);
  float* bt1=A32(NSEQ), *bt2=A32(NSEQ), *bt3=A32(NSEQ);
  float* X   = A32(NX);
  float* T2  = A32(NX);
  float* SUM = A32(NX);
  bf16_t* Xb   = A16(NX);
  bf16_t* Ab   = A16(NX);
  bf16_t* Qb16 = A16(NX), *Kb16 = A16(NX), *Vb16 = A16(NX);
  bf16_t* H1b  = A16(8LL*1024*2048);
  bf16_t *wq16[4], *wk16[4], *wv16[4], *wc16[4], *f116[4], *f216[4];
  for (int i = 0; i < 4; ++i){
    wq16[i]=A16(512LL*512); wk16[i]=A16(512LL*512);
    wv16[i]=A16(512LL*512); wc16[i]=A16(512LL*512);
    f116[i]=A16(2048LL*512); f216[i]=A16(512LL*2048);
  }

  auto lin = [&](const float* Xi, const float* Wi, const float* Bi, float* Yo,
                 long long R, int Kd, int Nd, int act){
    k_linear<<<nblk(R*Nd,256),256,0,stream>>>(Xi, Wi, Bi, Yo, R, Kd, Nd, act);
  };
  auto cvt = [&](const float* s, bf16_t* d, long long n){
    k_f2b<<<nblk(n,256),256,0,stream>>>(s, d, n);
  };
  auto gemmF = [&](const bf16_t* Ai, const bf16_t* Wi, const float* Bi, float* Ci,
                   int M, int N, int Kd, int act){
    long long threads = (long long)(M/16)*(N/64)*32;
    k_gemm_bf16<<<nblk(threads,256),256,0,stream>>>(Ai, Wi, Bi, Ci, nullptr, M, N, Kd, act);
  };
  auto gemmB = [&](const bf16_t* Ai, const bf16_t* Wi, const float* Bi, bf16_t* Ci,
                   int M, int N, int Kd, int act){
    long long threads = (long long)(M/16)*(N/64)*32;
    k_gemm_bf16<<<nblk(threads,256),256,0,stream>>>(Ai, Wi, Bi, nullptr, Ci, M, N, Kd, act);
  };
  auto attn = [&](const float* Qp, const float* Kp, const float* Vp, float* Op,
                  int Lq, int Lk, int nh, int D){
    long long tot = 8LL*nh*Lq;
    int hd = D/nh;
    float sc = 1.0f/sqrtf((float)hd);
    switch (hd){
      case 6:  k_attn<6 ><<<nblk(tot,128),128,0,stream>>>(Qp,Kp,Vp,Op,8,Lq,Lk,nh,D,sc); break;
      case 12: k_attn<12><<<nblk(tot,128),128,0,stream>>>(Qp,Kp,Vp,Op,8,Lq,Lk,nh,D,sc); break;
      default: k_attn<36><<<nblk(tot,128),128,0,stream>>>(Qp,Kp,Vp,Op,8,Lq,Lk,nh,D,sc); break;
    }
  };
  auto run_mha = [&](const float* qin, const float* kvin, int Lq, int Lk, int nh,
                     const MhaP& P, float* out){
    lin(qin,  P.iw,         P.ib,      bq, 8LL*Lq, 36, 36, 0);
    lin(kvin, P.iw + 36*36, P.ib + 36, bk, 8LL*Lk, 36, 36, 0);
    lin(kvin, P.iw + 72*36, P.ib + 72, bv, 8LL*Lk, 36, 36, 0);
    attn(bq, bk, bv, batt, Lq, Lk, nh, 36);
    lin(batt, P.ow, P.ob, out, 8LL*Lq, 36, 36, 0);
  };

  // ---- conv front-end ----
  k_conv1 <<<nblk(8LL*72*4096,256),256,0,stream>>>(x, c1_w, c1_b, bufA);
  k_c2pool<<<nblk(8LL*72*2048,256),256,0,stream>>>(bufA, c2_w, c2_b, bn1_g, bn1_b, bn1_m, bn1_v, bufB);
  k_c3    <<<nblk(8LL*36*2048,256),256,0,stream>>>(bufB, c3_w, c3_b, bufC);
  k_c4pool<<<nblk(8LL*36*1024,256),256,0,stream>>>(bufC, c4_w, c4_b, bn2_g, bn2_b, bn2_m, bn2_v, bufD);
  k_c5t   <<<nblk(8LL*1024*36,256),256,0,stream>>>(bufD, c5_w, c5_b, hseq);

  // ---- BiLSTM ----
  lin(hseq, wih_f, bih_f, preF, 8192, 36, 72, 0);
  lin(hseq, wih_b, bih_b, preR, 8192, 36, 72, 0);
  k_lstm<<<8,72,0,stream>>>(preF, whh_f, bhh_f, seq, 1024, 0);
  k_lstm<<<8,72,0,stream>>>(preR, whh_b, bhh_b, seq, 1024, 1);

  // ---- 6x DSTA ----
  for (int i = 0; i < 6; ++i){
    run_mha(seq, seq, 1024, 1024, 6, dl[i], bLarge);
    k_avgpool<<<nblk(8LL*512*36,256),256,0,stream>>>(seq, bxp, 8, 1024, 36, 2);
    run_mha(bxp, bxp, 512, 512, 3, dsm[i], bSmall);
    k_avgpool<<<nblk(8LL*256*36,256),256,0,stream>>>(seq, bxp, 8, 1024, 36, 4);
    run_mha(bxp, bxp, 256, 256, 1, dth[i], bThird);
    k_up2<<<nblk(8LL*1024*36,256),256,0,stream>>>(bSmall, bsu, 8, 512, 36);
    run_mha(bLarge, bsu, 1024, 1024, 6, dc1[i], bt1);
    k_add<<<nblk(NSEQ,256),256,0,stream>>>(bt1, bLarge, bt2, NSEQ);
    k_up2<<<nblk(8LL*512*36,256),256,0,stream>>>(bThird, bsu, 8, 256, 36);
    run_mha(bt2, bsu, 1024, 512, 6, dc2[i], bt1);
    k_add<<<nblk(NSEQ,256),256,0,stream>>>(bt1, bt2, bt3, NSEQ);
    k_add<<<nblk(NSEQ,256),256,0,stream>>>(seq, bt3, bt1, NSEQ);
    k_ln<<<nblk(8192,256),256,0,stream>>>(bt1, dg[i], db[i], seq, 8192, 36, 1e-5f);
  }

  // ---- proj + emb + positional encoding ----
  lin(seq, proj_w, proj_b, bt1, 8192, 36, 36, 0);
  lin(bt1, emb_w, emb_b, X, 8192, 36, 512, 0);
  k_posenc<<<nblk(NX,256),256,0,stream>>>(X, 1024, 512);

  // ---- convert encoder weights to bf16 ----
  for (int i = 0; i < 4; ++i){
    cvt(lay[i].wq, wq16[i], 512LL*512); cvt(lay[i].wk, wk16[i], 512LL*512);
    cvt(lay[i].wv, wv16[i], 512LL*512); cvt(lay[i].wc, wc16[i], 512LL*512);
    cvt(lay[i].f1, f116[i], 2048LL*512); cvt(lay[i].f2, f216[i], 512LL*2048);
  }

  // ---- 4x transformer encoder layers (WMMA GEMMs + WMMA flash attention) ----
  for (int i = 0; i < 4; ++i){
    cvt(X, Xb, NX);
    gemmB(Xb, wq16[i], lay[i].wqb, Qb16, 8192, 512, 512, 0);
    gemmB(Xb, wk16[i], lay[i].wkb, Kb16, 8192, 512, 512, 0);
    gemmB(Xb, wv16[i], lay[i].wvb, Vb16, 8192, 512, 512, 0);
    k_fattn<<<512,256,0,stream>>>(Qb16, Kb16, Vb16, Ab, 0.125f);
    gemmF(Ab, wc16[i], lay[i].wcb, T2, 8192, 512, 512, 0);
    k_add<<<nblk(NX,256),256,0,stream>>>(X, T2, SUM, NX);
    k_ln<<<nblk(8192,256),256,0,stream>>>(SUM, lay[i].n1g, lay[i].n1b, X, 8192, 512, 1e-12f);
    cvt(X, Xb, NX);
    gemmB(Xb, f116[i], lay[i].f1b, H1b, 8192, 2048, 512, 1);   // fused exact GELU, bf16 out
    gemmF(H1b, f216[i], lay[i].f2bb, T2, 8192, 512, 2048, 0);
    k_add<<<nblk(NX,256),256,0,stream>>>(X, T2, SUM, NX);
    k_ln<<<nblk(8192,256),256,0,stream>>>(SUM, lay[i].n2g, lay[i].n2b, X, 8192, 512, 1e-12f);
  }

  k_copy<<<nblk(NX,256),256,0,stream>>>(X, (float*)d_out, NX);
}
